// QuantumAttention_50405736186430
// MI455X (gfx1250) — matrix operand order repaired and checked by reading, not compiled, not run on hardware
//
#include <hip/hip_runtime.h>
#include <math.h>

// ---------------------------------------------------------------------------
// QuantumAttention for MI455X (gfx1250), wave32 + WMMA f16 (f32 accumulate).
// B=2, S=2048, HID=1024, NH=16, D=64. Quantum transforms folded into weights.
// ---------------------------------------------------------------------------

typedef __attribute__((ext_vector_type(16))) _Float16 v16h;
typedef __attribute__((ext_vector_type(8)))  _Float16 v8h;
typedef __attribute__((ext_vector_type(4)))  _Float16 v4h;
typedef __attribute__((ext_vector_type(8)))  float    v8f;

#define R_SQRT2 0.70710678118654752440f
#define ATTN_SCALE 0.125f   /* 1/sqrt(64) */

// ---- gfx1250 async global->LDS staging (ASYNCcnt-tracked), with fallback ----
#if defined(__has_builtin)
#if __has_builtin(__builtin_amdgcn_global_load_async_to_lds_b128) && \
    __has_builtin(__builtin_amdgcn_s_wait_asynccnt)
#define USE_ASYNC_LDS 1
#endif
#endif

#ifdef USE_ASYNC_LDS
typedef int v4i_128 __attribute__((vector_size(16)));
typedef __attribute__((address_space(1))) void     gas_void;
typedef __attribute__((address_space(3))) void     las_void;
typedef __attribute__((address_space(1))) v4i_128* gas_b128_t;
typedef __attribute__((address_space(3))) v4i_128* las_b128_t;
__device__ __forceinline__ void stage_b128(const void* gsrc, void* ldst) {
  // 16B per lane, memory -> LDS, no VGPR bounce; tracked by ASYNCcnt
  __builtin_amdgcn_global_load_async_to_lds_b128(
      (gas_b128_t)(gas_void*)gsrc, (las_b128_t)(las_void*)ldst, 0, 0);
}
__device__ __forceinline__ void stage_wait() {
  __builtin_amdgcn_s_wait_asynccnt(0);
}
#else
__device__ __forceinline__ void stage_b128(const void* gsrc, void* ldst) {
  *(v8h*)ldst = *(const v8h*)gsrc;
}
__device__ __forceinline__ void stage_wait() {}
#endif

__device__ __forceinline__ v16h frag_cat(v8h lo, v8h hi) {
  return __builtin_shufflevector(lo, hi, 0,1,2,3,4,5,6,7,8,9,10,11,12,13,14,15);
}
// A fragment (16x32 f16): lane row = l&15; halves at kb_a=(l<16?0:8) and kb_a+16
__device__ __forceinline__ v16h load_a16(const _Float16* rowp, int kb_a) {
  v8h lo = *(const v8h*)(rowp + kb_a);
  v8h hi = *(const v8h*)(rowp + kb_a + 16);
  return frag_cat(lo, hi);
}
// B fragment (32x16 f16): lane col = l&15; halves at kb_b/2=(l<16?0:8) and kb_b/2+16
__device__ __forceinline__ v16h load_b16(const _Float16* rowp, int kb_b) {
  v8h lo = *(const v8h*)(rowp + (kb_b >> 1));
  v8h hi = *(const v8h*)(rowp + (kb_b >> 1) + 16);
  return frag_cat(lo, hi);
}
__device__ __forceinline__ v8f wmma16(v16h a, v16h b, v8f c) {
  return __builtin_amdgcn_wmma_f32_16x16x32_f16(false, a, false, b, (short)0, c,
                                                false, false);
}
__device__ __forceinline__ v8f v8f_zero() {
  v8f z = {0.f,0.f,0.f,0.f,0.f,0.f,0.f,0.f};
  return z;
}

// ---------------------------------------------------------------------------
// Kernel 0: build the fused 64x64 head transforms T_qk, T_v.
//   x -> entangle(ent_w) -> CNOT perm -> *ent_s -> Hadamard -> *s -> {Hadamard | CNOT}
// All are right-multiplications on the 64-dim head vector; compose columns.
// ---------------------------------------------------------------------------
__global__ __launch_bounds__(64)
void build_T_kernel(const float* __restrict__ ent_w, const float* __restrict__ ent_s,
                    const float* __restrict__ sup_w, const float* __restrict__ sup_c,
                    float* __restrict__ Tqk, float* __restrict__ Tv) {
  int r = threadIdx.x;
  if (r >= 64) return;
  int p = r >> 4, i = r & 15;
  float row[64];
  #pragma unroll
  for (int c = 0; c < 64; ++c)
    row[c] = ((c >> 4) == p) ? ent_w[p * 256 + i * 16 + (c & 15)] : 0.f;
  // CNOT on columns: within each group of 4, swap cols 2 and 3
  #pragma unroll
  for (int g = 0; g < 16; ++g) {
    float tv = row[g*4+2]; row[g*4+2] = row[g*4+3]; row[g*4+3] = tv;
  }
  #pragma unroll
  for (int c = 0; c < 64; ++c) row[c] *= ent_s[c >> 4];
  // Hadamard on column pairs
  #pragma unroll
  for (int t = 0; t < 32; ++t) {
    float a = row[2*t], b = row[2*t+1];
    row[2*t] = (a + b) * R_SQRT2; row[2*t+1] = (a - b) * R_SQRT2;
  }
  // superposition diagonal: s[c] = sum_n sup_c[n]*sup_w[n][c]
  #pragma unroll
  for (int c = 0; c < 64; ++c) {
    float s = 0.f;
    #pragma unroll
    for (int n = 0; n < 8; ++n) s += sup_c[n] * sup_w[n * 64 + c];
    row[c] *= s;
  }
  // Tqk: one more Hadamard
  #pragma unroll
  for (int t = 0; t < 32; ++t) {
    float a = row[2*t], b = row[2*t+1];
    Tqk[r*64 + 2*t]     = (a + b) * R_SQRT2;
    Tqk[r*64 + 2*t + 1] = (a - b) * R_SQRT2;
  }
  // Tv: one more CNOT column permutation
  #pragma unroll
  for (int g = 0; g < 16; ++g) {
    Tv[r*64 + g*4 + 0] = row[g*4+0];
    Tv[r*64 + g*4 + 1] = row[g*4+1];
    Tv[r*64 + g*4 + 2] = row[g*4+3];
    Tv[r*64 + g*4 + 3] = row[g*4+2];
  }
}

// ---------------------------------------------------------------------------
// Kernel 1: effective f16 weights. z=0:Wq(Tqk) 1:Wk(Tqk) 2:Wv(Tv) 3:Wo(identity)
// Output row j = output feature (already transposed for B-fragment loads):
//   Weff[h*64+d][k] = sum_{d'} W[h*64+d'][k] * T[d'][d]
// grid (4 kblocks, 16 heads, 4 z), 256 threads. Dyn LDS: W tile + T.
// ---------------------------------------------------------------------------
__global__ __launch_bounds__(256)
void build_weights_kernel(const float* __restrict__ Wq, const float* __restrict__ Wk,
                          const float* __restrict__ Wv, const float* __restrict__ Wo,
                          const float* __restrict__ Tqk, const float* __restrict__ Tv,
                          _Float16* __restrict__ Wout) {
  extern __shared__ float smw[];
  float* Wld = smw;            // [64][260]
  float* Tld = smw + 64 * 260; // [4096]
  int z = blockIdx.z, h = blockIdx.y;
  int kg = blockIdx.x * 256 + threadIdx.x;
  int tx = threadIdx.x;
  const float* W = (z == 0) ? Wq : (z == 1) ? Wk : (z == 2) ? Wv : Wo;
  _Float16* out = Wout + (size_t)z * 1048576;
  if (z == 3) {  // plain f16 convert of Wo (already row = output feature)
    #pragma unroll 1
    for (int d = 0; d < 64; ++d) {
      size_t j = (size_t)(h * 64 + d);
      out[j * 1024 + kg] = (_Float16)W[j * 1024 + kg];
    }
    return;
  }
  const float* T = (z == 2) ? Tv : Tqk;
  for (int idx = tx; idx < 4096; idx += 256) Tld[idx] = T[idx];
  #pragma unroll 1
  for (int dp = 0; dp < 64; ++dp)
    Wld[dp * 260 + tx] = W[(size_t)(h * 64 + dp) * 1024 + kg];
  __syncthreads();
  float acc[64];
  #pragma unroll
  for (int d = 0; d < 64; ++d) acc[d] = 0.f;
  #pragma unroll 1
  for (int dp = 0; dp < 64; ++dp) {
    float wv = Wld[dp * 260 + tx];
    #pragma unroll
    for (int d = 0; d < 64; ++d) acc[d] += wv * Tld[dp * 64 + d];
  }
  #pragma unroll 1
  for (int d = 0; d < 64; ++d)
    out[(size_t)(h * 64 + d) * 1024 + kg] = (_Float16)acc[d];
}

// Effective biases: beff[z][h*64+d] = sum_{d'} b[h*64+d'] * T[d'][d]
__global__ __launch_bounds__(64)
void build_bias_kernel(const float* __restrict__ bq, const float* __restrict__ bk,
                       const float* __restrict__ bv,
                       const float* __restrict__ Tqk, const float* __restrict__ Tv,
                       float* __restrict__ beff) {
  int z = blockIdx.x;
  const float* b = (z == 0) ? bq : (z == 1) ? bk : bv;
  const float* T = (z == 2) ? Tv : Tqk;
  int d = threadIdx.x;
  for (int h = 0; h < 16; ++h) {
    float acc = 0.f;
    #pragma unroll 4
    for (int dp = 0; dp < 64; ++dp) acc += b[h * 64 + dp] * T[dp * 64 + d];
    beff[z * 1024 + h * 64 + d] = acc;
  }
}

// ---------------------------------------------------------------------------
// WMMA GEMM: C[4096x1024] = A[4096x1024] * Bt^T + bias   (Bt row = out feature)
// BM=BN=128, BK=32, 256 threads = 8 waves in a 2x4 wave grid; each wave 64x32.
// AF32: A is f32 (converted to f16 while staging); else A is f16 (async->LDS).
// OUTF16: C stored as f16 (ws) vs f32 (final output).
// ---------------------------------------------------------------------------
template <bool AF32, bool OUTF16>
__global__ __launch_bounds__(256)
void gemm_kernel(const void* __restrict__ Aptr, const _Float16* __restrict__ Bt,
                 const float* __restrict__ bias, void* __restrict__ Cptr) {
  __shared__ _Float16 As[128 * 40];
  __shared__ _Float16 Bs[128 * 40];
  __shared__ float biasLd[128];
  const int t = threadIdx.x, lane = t & 31, w = t >> 5;
  const int rl = lane & 15;
  const int kb_a = (lane < 16) ? 0 : 8;
  const int kb_b = (lane < 16) ? 0 : 16;
  const int m0 = blockIdx.y * 128, n0 = blockIdx.x * 128;
  const int wm = w >> 2, wn = w & 3;

  if (t < 128) biasLd[t] = bias[n0 + t];

  v8f acc[4][2];
  #pragma unroll
  for (int mt = 0; mt < 4; ++mt)
    #pragma unroll
    for (int nt = 0; nt < 2; ++nt) acc[mt][nt] = v8f_zero();

  for (int k0 = 0; k0 < 1024; k0 += 32) {
    __syncthreads();
    // ---- stage A tile (128x32) ----
    if (AF32) {
      const float* A = (const float*)Aptr;
      #pragma unroll
      for (int it = 0; it < 4; ++it) {
        int pos = it * 256 + t;                 // 1024 float4 positions
        int r = pos >> 3, c = (pos & 7) * 4;
        float4 f = *(const float4*)(A + (size_t)(m0 + r) * 1024 + k0 + c);
        v4h hv; hv[0] = (_Float16)f.x; hv[1] = (_Float16)f.y;
        hv[2] = (_Float16)f.z; hv[3] = (_Float16)f.w;
        *(v4h*)&As[r * 40 + c] = hv;
      }
      if (k0 + 32 < 1024)
        __builtin_prefetch((const char*)((const float*)Aptr +
                           (size_t)(m0 + (t & 127)) * 1024 + k0 + 32), 0, 1);
    } else {
      const _Float16* A = (const _Float16*)Aptr;
      #pragma unroll
      for (int it = 0; it < 2; ++it) {
        int pos = it * 256 + t;                 // 512 v8h chunks
        int r = pos >> 2, c = (pos & 3) * 8;
        stage_b128(A + (size_t)(m0 + r) * 1024 + k0 + c, &As[r * 40 + c]);
      }
      if (k0 + 32 < 1024)
        __builtin_prefetch((const char*)((const _Float16*)Aptr +
                           (size_t)(m0 + (t & 127)) * 1024 + k0 + 32), 0, 1);
    }
    // ---- stage B tile (128 out-features x 32 k), async memory->LDS ----
    #pragma unroll
    for (int it = 0; it < 2; ++it) {
      int pos = it * 256 + t;
      int r = pos >> 2, c = (pos & 3) * 8;
      stage_b128(Bt + (size_t)(n0 + r) * 1024 + k0 + c, &Bs[r * 40 + c]);
    }
    if (k0 + 32 < 1024)
      __builtin_prefetch((const char*)(Bt + (size_t)(n0 + (t & 127)) * 1024 + k0 + 32),
                         0, 1);
    stage_wait();          // ASYNCcnt -> 0 before publishing LDS
    __syncthreads();
    // ---- compute: one K=32 WMMA step per tile ----
    v16h bfr[2];
    #pragma unroll
    for (int nt = 0; nt < 2; ++nt)
      bfr[nt] = load_b16(&Bs[(wn * 32 + nt * 16 + rl) * 40], kb_b);
    #pragma unroll
    for (int mt = 0; mt < 4; ++mt) {
      v16h af = load_a16(&As[(wm * 64 + mt * 16 + rl) * 40], kb_a);
      #pragma unroll
      for (int nt = 0; nt < 2; ++nt)
        acc[mt][nt] = wmma16(af, bfr[nt], acc[mt][nt]);
    }
  }
  // ---- epilogue: bias add + store ----
  #pragma unroll
  for (int mt = 0; mt < 4; ++mt)
    #pragma unroll
    for (int nt = 0; nt < 2; ++nt)
      #pragma unroll
      for (int v = 0; v < 8; ++v) {
        int row = m0 + wm * 64 + mt * 16 + v + ((lane < 16) ? 0 : 8);
        int col = wn * 32 + nt * 16 + rl;
        float val = acc[mt][nt][v] + biasLd[col];
        if (OUTF16)
          ((_Float16*)Cptr)[(size_t)row * 1024 + n0 + col] = (_Float16)val;
        else
          ((float*)Cptr)[(size_t)row * 1024 + n0 + col] = val;
      }
}

// ---------------------------------------------------------------------------
// Attention kernel: per (b,h) and 64-row q-tile. Flash-style context with
// raw scores streamed to the attn output region plus per-row (m,l) stats.
// Dynamic LDS carve (bytes):
//   Ql   [64][72]  f16   @0      (9216)
//   Kl   [128][72] f16   @9216   (18432)
//   Pl   [64][136] f16   @27648  (17408)
//   Vt   [64][136] f16   @45056  (17408)
//   Sld  [64][128] f32   @62464  (32768)
//   mrow/lrow/rfrow [64] @95232  (768)
//   pm/pl [64][4]  f32   @96000  (2048)   total 98048
// ---------------------------------------------------------------------------
#define ATTN_LDS_BYTES 98048

__global__ __launch_bounds__(256)
void attn_kernel(const _Float16* __restrict__ Qq, const _Float16* __restrict__ Kq,
                 const _Float16* __restrict__ Vq, float* __restrict__ attnOut,
                 _Float16* __restrict__ ctxOut, float* __restrict__ statsM,
                 float* __restrict__ statsL) {
  extern __shared__ char smem[];
  _Float16* Ql = (_Float16*)(smem);
  _Float16* Kl = (_Float16*)(smem + 9216);
  _Float16* Pl = (_Float16*)(smem + 27648);
  _Float16* Vt = (_Float16*)(smem + 45056);
  float* Sld   = (float*)(smem + 62464);
  float* mrow  = (float*)(smem + 95232);
  float* lrow  = mrow + 64;
  float* rfrow = mrow + 128;
  float* pm    = (float*)(smem + 96000);   // [64][4]
  float* pl    = pm + 256;

  const int t = threadIdx.x, lane = t & 31, w = t >> 5;
  const int rl = lane & 15;
  const int kb_a = (lane < 16) ? 0 : 8;
  const int kb_b = (lane < 16) ? 0 : 16;
  const int rofs = (lane < 16) ? 0 : 8;
  const int bh = blockIdx.y;             // 0..31
  const int b = bh >> 4, h = bh & 15;
  const int q0 = blockIdx.x * 64;

  const size_t baseQ = ((size_t)(b * 2048 + q0)) * 1024 + h * 64;
  float* attnBase = attnOut + ((size_t)bh * 2048 + q0) * 2048;

  // stage Q tile [64 rows][64 d] once (async memory->LDS)
  #pragma unroll
  for (int it = 0; it < 2; ++it) {
    int pos = it * 256 + t;               // 512 v8h chunks
    int r = pos >> 3, c = (pos & 7) * 8;
    stage_b128(Qq + baseQ + (size_t)r * 1024 + c, &Ql[r * 72 + c]);
  }
  if (t < 64) { mrow[t] = -__builtin_inff(); lrow[t] = 0.f; }

  // context accumulators: 4(m) x 4(n) tiles over 64x64, 2 per wave
  const int mi = w & 3;
  const int nb = (w >> 2) * 2;
  v8f ctx[2];
  ctx[0] = v8f_zero(); ctx[1] = v8f_zero();

  for (int s0 = 0; s0 < 2048; s0 += 128) {
    __syncthreads();                                   // (1) LDS reuse
    const size_t baseKV = ((size_t)(b * 2048 + s0)) * 1024 + h * 64;
    // ---- stage K tile [128 keys][64 d] (async memory->LDS) ----
    #pragma unroll
    for (int it = 0; it < 4; ++it) {
      int pos = it * 256 + t;             // 1024 v8h chunks
      int r = pos >> 3, c = (pos & 7) * 8;
      stage_b128(Kq + baseKV + (size_t)r * 1024 + c, &Kl[r * 72 + c]);
    }
    stage_wait();                                      // Q (iter 0) + K complete
    __syncthreads();                                   // (2) Kl ready
    // ---- scores: wave w owns col-tile w (16 cols), all 4 row-tiles ----
    v8f sacc[4];
    #pragma unroll
    for (int mt = 0; mt < 4; ++mt) sacc[mt] = v8f_zero();
    #pragma unroll
    for (int k32 = 0; k32 < 2; ++k32) {
      v16h bfr = load_b16(&Kl[(w * 16 + rl) * 72 + k32 * 32], kb_b);
      #pragma unroll
      for (int mt = 0; mt < 4; ++mt) {
        v16h af = load_a16(&Ql[(mt * 16 + rl) * 72 + k32 * 32], kb_a);
        sacc[mt] = wmma16(af, bfr, sacc[mt]);
      }
    }
    #pragma unroll
    for (int mt = 0; mt < 4; ++mt)
      #pragma unroll
      for (int v = 0; v < 8; ++v)
        Sld[(mt * 16 + v + rofs) * 128 + w * 16 + rl] = sacc[mt][v] * ATTN_SCALE;
    __syncthreads();                                   // (3) Sld ready
    // ---- per-row partial max / sumexp (thread = row r, quarter qd) ----
    {
      int r = t >> 2, qd = t & 3;
      const float* Sr = &Sld[r * 128 + qd * 32];
      float tm = -__builtin_inff();
      #pragma unroll 8
      for (int i = 0; i < 32; ++i) tm = fmaxf(tm, Sr[i]);
      float ts = 0.f;
      #pragma unroll 8
      for (int i = 0; i < 32; ++i) ts += __expf(Sr[i] - tm);
      pm[r * 4 + qd] = tm; pl[r * 4 + qd] = ts;
    }
    __syncthreads();                                   // (4) partials ready
    if (t < 64) {
      float M = fmaxf(fmaxf(pm[t*4+0], pm[t*4+1]), fmaxf(pm[t*4+2], pm[t*4+3]));
      float L = pl[t*4+0] * __expf(pm[t*4+0] - M) + pl[t*4+1] * __expf(pm[t*4+1] - M)
              + pl[t*4+2] * __expf(pm[t*4+2] - M) + pl[t*4+3] * __expf(pm[t*4+3] - M);
      float mo = mrow[t];
      float mn = fmaxf(mo, M);
      float rf = __expf(mo - mn);
      lrow[t] = lrow[t] * rf + L * __expf(M - mn);
      mrow[t] = mn;
      rfrow[t] = rf;
    }
    __syncthreads();                                   // (5) stats ready
    // ---- write raw scores + build P = exp(s - m) tile ----
    #pragma unroll 4
    for (int it = 0; it < 32; ++it) {
      int pos = it * 256 + t;             // 8192 elements, coalesced rows
      int r = pos >> 7, c = pos & 127;
      float sv = Sld[r * 128 + c];
      attnBase[(size_t)r * 2048 + s0 + c] = sv;
      Pl[r * 136 + c] = (_Float16)__expf(sv - mrow[r]);
    }
    // ---- stage V tile transposed: Vt[d][s_local] (manual: transposes) ----
    #pragma unroll
    for (int it = 0; it < 4; ++it) {
      int pos = it * 256 + t;
      int r = pos >> 3, c = (pos & 7) * 8;
      v8h v = *(const v8h*)(Vq + baseKV + (size_t)r * 1024 + c);
      #pragma unroll
      for (int j = 0; j < 8; ++j) Vt[(c + j) * 136 + r] = v[j];
    }
    // ---- rescale running context by exp(m_old - m_new) ----
    #pragma unroll
    for (int j = 0; j < 2; ++j)
      #pragma unroll
      for (int v = 0; v < 8; ++v)
        ctx[j][v] *= rfrow[mi * 16 + v + rofs];
    __syncthreads();                                   // (6) Pl, Vt ready
    // ---- context WMMA: ctx += P(64x128) x V(128x64) ----
    #pragma unroll
    for (int k32 = 0; k32 < 4; ++k32) {
      v16h af = load_a16(&Pl[(mi * 16 + rl) * 136 + k32 * 32], kb_a);
      #pragma unroll
      for (int j = 0; j < 2; ++j) {
        v16h bfr = load_b16(&Vt[((nb + j) * 16 + rl) * 136 + k32 * 32], kb_b);
        ctx[j] = wmma16(af, bfr, ctx[j]);
      }
    }
  }
  // ---- finalize: ctx /= l, store f16 ----
  #pragma unroll
  for (int j = 0; j < 2; ++j)
    #pragma unroll
    for (int v = 0; v < 8; ++v) {
      int row = mi * 16 + v + rofs;
      int col = (nb + j) * 16 + rl;
      float invl = 1.f / lrow[row];
      ctxOut[((size_t)(b * 2048 + q0 + row)) * 1024 + h * 64 + col] =
          (_Float16)(ctx[j][v] * invl);
    }
  if (t < 64) {
    size_t sidx = (size_t)bh * 2048 + q0 + t;
    statsM[sidx] = mrow[t];
    statsL[sidx] = lrow[t];
  }
}

// Normalize raw scores into final attn: p = exp(s - m) / l. One block per row.
__global__ __launch_bounds__(256)
void norm_attn_kernel(float* __restrict__ attn, const float* __restrict__ statsM,
                      const float* __restrict__ statsL) {
  size_t row = blockIdx.x;
  float m = statsM[row];
  float inv = 1.f / statsL[row];
  float4* p = (float4*)(attn + row * 2048);
  #pragma unroll
  for (int i = 0; i < 2; ++i) {
    int idx = i * 256 + threadIdx.x;      // 512 float4 per row
    float4 v = p[idx];
    v.x = __expf(v.x - m) * inv;
    v.y = __expf(v.y - m) * inv;
    v.z = __expf(v.z - m) * inv;
    v.w = __expf(v.w - m) * inv;
    p[idx] = v;
  }
}

// ---------------------------------------------------------------------------
// Launcher
// ---------------------------------------------------------------------------
extern "C" void kernel_launch(void* const* d_in, const int* in_sizes, int n_in,
                              void* d_out, int out_size, void* d_ws, size_t ws_size,
                              hipStream_t stream) {
  (void)in_sizes; (void)n_in; (void)out_size; (void)ws_size;
  const float* query = (const float*)d_in[0];
  const float* key   = (const float*)d_in[1];
  const float* value = (const float*)d_in[2];
  const float* Wq = (const float*)d_in[3];  const float* bq = (const float*)d_in[4];
  const float* Wk = (const float*)d_in[5];  const float* bk = (const float*)d_in[6];
  const float* Wv = (const float*)d_in[7];  const float* bv = (const float*)d_in[8];
  const float* Wo = (const float*)d_in[9];  const float* bo = (const float*)d_in[10];
  const float* ent_w = (const float*)d_in[11];
  const float* ent_s = (const float*)d_in[12];
  const float* sup_w = (const float*)d_in[13];
  const float* sup_c = (const float*)d_in[14];

  // workspace carve: f32 region then f16 region
  float* wsF   = (float*)d_ws;
  float* Tqk   = wsF;                 // 4096
  float* Tv    = wsF + 4096;          // 4096
  float* beff  = wsF + 8192;          // 3*1024
  float* statsM = wsF + 11264;        // 65536
  float* statsL = statsM + 65536;     // 65536  -> f32 end at 142336 floats
  _Float16* wsH = (_Float16*)((char*)d_ws + 142336u * sizeof(float));
  _Float16* Weff = wsH;               // 4 x 1024*1024 (q,k,v,o)
  _Float16* Qq = wsH + 4u * 1048576u; // 4096x1024
  _Float16* Kq = Qq + 4194304u;
  _Float16* Vq = Kq + 4194304u;
  _Float16* Ctx = Vq + 4194304u;

  float* outPtr  = (float*)d_out;                 // [2,2048,1024]
  float* attnPtr = outPtr + 4194304u;             // [2,16,2048,2048]

  build_T_kernel<<<dim3(1), dim3(64), 0, stream>>>(ent_w, ent_s, sup_w, sup_c,
                                                   Tqk, Tv);
  build_weights_kernel<<<dim3(4, 16, 4), dim3(256),
                         (64 * 260 + 4096) * sizeof(float), stream>>>(
      Wq, Wk, Wv, Wo, Tqk, Tv, Weff);
  build_bias_kernel<<<dim3(3), dim3(64), 0, stream>>>(bq, bk, bv, Tqk, Tv, beff);

  gemm_kernel<true, true><<<dim3(8, 32), dim3(256), 0, stream>>>(
      (const void*)query, Weff, beff, (void*)Qq);
  gemm_kernel<true, true><<<dim3(8, 32), dim3(256), 0, stream>>>(
      (const void*)key, Weff + 1048576u, beff + 1024, (void*)Kq);
  gemm_kernel<true, true><<<dim3(8, 32), dim3(256), 0, stream>>>(
      (const void*)value, Weff + 2u * 1048576u, beff + 2048, (void*)Vq);

  attn_kernel<<<dim3(32, 32), dim3(256), ATTN_LDS_BYTES, stream>>>(
      Qq, Kq, Vq, attnPtr, Ctx, statsM, statsL);

  norm_attn_kernel<<<dim3(65536), dim3(256), 0, stream>>>(attnPtr, statsM, statsL);

  gemm_kernel<false, false><<<dim3(8, 32), dim3(256), 0, stream>>>(
      (const void*)Ctx, Weff + 3u * 1048576u, bo, (void*)outPtr);
}
